// FFTInspiredAttention_61143154426287
// MI455X (gfx1250) — matrix operand order repaired and checked by reading, not compiled, not run on hardware
//
#include <hip/hip_runtime.h>
#include <hip/hip_bf16.h>
#include <stdint.h>

// Problem constants (from reference)
#define B_     4
#define N_     4096
#define D_     512
#define H_     8
#define DH_    64
#define INNER_ 512
#define LOGN_  12
#define SCALE_ 0.125f          // DH^-0.5
#define BN_    (B_ * N_)       // 16384 rows

typedef __bf16          v16bf __attribute__((ext_vector_type(16)));
typedef float           v8f   __attribute__((ext_vector_type(8)));
typedef unsigned int    u32x4 __attribute__((ext_vector_type(4)));
typedef float           f32x4 __attribute__((ext_vector_type(4)));
typedef unsigned short  u16x4 __attribute__((ext_vector_type(4)));
typedef unsigned short  u16x8 __attribute__((ext_vector_type(8)));

__device__ __forceinline__ unsigned short f2bf(float f) {
  // round-to-nearest-even fp32 -> bf16
  unsigned int u = __float_as_uint(f);
  u += 0x7FFFu + ((u >> 16) & 1u);
  return (unsigned short)(u >> 16);
}

__device__ __forceinline__ float bf2f(unsigned short b) {
  return __uint_as_float((unsigned int)b << 16);
}

union FragBf { v16bf v; u32x4 q[2]; };

// ---------------------------------------------------------------------------
// fp32 -> bf16 bulk convert, 4 elements / thread
// ---------------------------------------------------------------------------
__global__ __launch_bounds__(256) void cvt_f32_bf16_x4(const float* __restrict__ src,
                                                       unsigned short* __restrict__ dst,
                                                       int n4) {
  int i = blockIdx.x * blockDim.x + threadIdx.x;
  if (i >= n4) return;
  f32x4 f = ((const f32x4*)src)[i];
  u16x4 o;
  o.x = f2bf(f.x); o.y = f2bf(f.y); o.z = f2bf(f.z); o.w = f2bf(f.w);
  ((u16x4*)dst)[i] = o;
}

// ---------------------------------------------------------------------------
// bf16 WMMA GEMM:  C[M,Nc] = A[M,K] * Bm[Nc,K]^T   (fp32 accumulate)
// Block = 256 threads = 8 wave32s arranged 4(M) x 2(N); block tile 128x128.
// Wave tile 32x64 = 2x4 fragments of v_wmma_f32_16x16x32_bf16.
// A/B fragments loaded straight from global (K-contiguous rows match the
// per-lane WMMA VGPR layouts; two global_load_b128 per fragment per lane).
// Optionally writes fp32 C and/or bf16 C (either pointer may be null).
// ---------------------------------------------------------------------------
__global__ __launch_bounds__(256) void gemm_bf16_wmma(const __bf16* __restrict__ A,
                                                      const __bf16* __restrict__ Bm,
                                                      float* __restrict__ Cf,
                                                      unsigned short* __restrict__ Cb,
                                                      int M, int Nc, int K) {
  const int lane  = threadIdx.x & 31;
  const int wid   = threadIdx.x >> 5;      // 0..7
  const int waveM = wid & 3;               // 4 waves along M
  const int waveN = wid >> 2;              // 2 waves along N
  const int m0 = blockIdx.y * 128 + waveM * 32;
  const int n0 = blockIdx.x * 128 + waveN * 64;
  const int r  = lane & 15;                // row/col within 16x16 fragment
  const int hi = lane >> 4;                // which K half-group this lane holds

  v8f acc[2][4];
#pragma unroll
  for (int i = 0; i < 2; ++i)
#pragma unroll
    for (int j = 0; j < 4; ++j)
      acc[i][j] = (v8f)0.0f;

  const __bf16* aRow[2];
#pragma unroll
  for (int i = 0; i < 2; ++i) aRow[i] = A + (size_t)(m0 + i * 16 + r) * K;
  const __bf16* bRow[4];
#pragma unroll
  for (int j = 0; j < 4; ++j) bRow[j] = Bm + (size_t)(n0 + j * 16 + r) * K;

  for (int k0 = 0; k0 < K; k0 += 32) {
    FragBf a[2], b[4];
    // A 16x32 bf16 layout: lanes 0-15 hold K=k0+[0..7] and k0+16+[0..7],
    //                      lanes 16-31 hold K=k0+8+[0..7] and k0+24+[0..7]
#pragma unroll
    for (int i = 0; i < 2; ++i) {
      const __bf16* p = aRow[i] + k0 + hi * 8;
      a[i].q[0] = *(const u32x4*)(p);
      a[i].q[1] = *(const u32x4*)(p + 16);
    }
    // B 32x16 bf16 layout: lanes 0-15 hold K=k0+[0..7] and k0+16+[0..7],
    //                      lanes 16-31 hold K=k0+8+[0..7] and k0+24+[0..7]
#pragma unroll
    for (int j = 0; j < 4; ++j) {
      const __bf16* p = bRow[j] + k0 + hi * 8;
      b[j].q[0] = *(const u32x4*)(p);
      b[j].q[1] = *(const u32x4*)(p + 16);
    }
#pragma unroll
    for (int i = 0; i < 2; ++i)
#pragma unroll
      for (int j = 0; j < 4; ++j)
        acc[i][j] = __builtin_amdgcn_wmma_f32_16x16x32_bf16(
            /*neg_a=*/false, a[i].v, /*neg_b=*/false, b[j].v,
            /*c_mod=*/(short)0, acc[i][j], /*reuse_a=*/false, /*reuse_b=*/false);
  }

  // C/D 16x16 f32 layout: VGPR t -> (M = t + 8*hi, N = lane&15)
  const int mOff = 8 * hi;
#pragma unroll
  for (int i = 0; i < 2; ++i) {
#pragma unroll
    for (int j = 0; j < 4; ++j) {
#pragma unroll
      for (int t = 0; t < 8; ++t) {
        int m = m0 + i * 16 + t + mOff;
        int n = n0 + j * 16 + r;
        size_t off = (size_t)m * Nc + n;
        float val = acc[i][j][t];
        if (Cf) Cf[off] = val;
        if (Cb) Cb[off] = f2bf(val);
      }
    }
  }
}

// ---------------------------------------------------------------------------
// Butterfly pair softmax + recombine.
// 8 threads handle one (batch, pair, head); each thread owns 8 of 64 channels.
// qk is stored bf16 (it came out of a bf16 GEMM anyway); dots accumulate fp32.
// da = q_a.k_a * scale, db = q_a.k_b * scale; 2-way softmax; h = w * (v_a+v_b).
// Writes bf16 h for the next stage, or fp32 to d_out on the final stage.
// ---------------------------------------------------------------------------
__global__ __launch_bounds__(256) void stage_pair_softmax(const unsigned short* __restrict__ qk,
                                                          const float* __restrict__ vf,
                                                          unsigned short* __restrict__ hout,
                                                          float* __restrict__ outf,
                                                          int stage) {
  const int gid   = blockIdx.x * 32 + (threadIdx.x >> 3);  // group id
  const int t     = threadIdx.x & 7;
  const int head  = gid & (H_ - 1);
  const int p     = (gid >> 3) & (N_ / 2 - 1);
  const int batch = gid >> 14;                             // 2048 pairs * 8 heads = 2^14
  const int stride = 1 << stage;
  const int a = ((p >> stage) << (stage + 1)) | (p & (stride - 1));
  const int b = a + stride;
  const size_t rowA = (size_t)batch * N_ + a;
  const size_t rowB = (size_t)batch * N_ + b;

  // one 16B load per operand per thread
  const u16x8 qa = *(const u16x8*)(qk + rowA * (2 * INNER_) + head * DH_ + t * 8);
  const u16x8 ka = *(const u16x8*)(qk + rowA * (2 * INNER_) + INNER_ + head * DH_ + t * 8);
  const u16x8 kb = *(const u16x8*)(qk + rowB * (2 * INNER_) + INNER_ + head * DH_ + t * 8);

  float da = 0.f, db = 0.f;
#pragma unroll
  for (int j = 0; j < 8; ++j) {
    float q = bf2f(qa[j]);
    da += q * bf2f(ka[j]);
    db += q * bf2f(kb[j]);
  }
#pragma unroll
  for (int off = 4; off; off >>= 1) {
    da += __shfl_xor(da, off, 8);
    db += __shfl_xor(db, off, 8);
  }
  da *= SCALE_;
  db *= SCALE_;
  float m  = fmaxf(da, db);
  float ea = __expf(da - m), eb = __expf(db - m);
  float inv = 1.0f / (ea + eb);
  float wa = ea * inv, wb = eb * inv;

  const int ch = head * DH_ + t * 8;
  const float* va = vf + rowA * INNER_ + ch;
  const float* vb = vf + rowB * INNER_ + ch;
  if (outf) {
#pragma unroll
    for (int j = 0; j < 8; ++j) {
      float vs = va[j] + vb[j];
      outf[rowA * INNER_ + ch + j] = wa * vs;
      outf[rowB * INNER_ + ch + j] = wb * vs;
    }
  } else {
#pragma unroll
    for (int j = 0; j < 8; ++j) {
      float vs = va[j] + vb[j];
      hout[rowA * INNER_ + ch + j] = f2bf(wa * vs);
      hout[rowB * INNER_ + ch + j] = f2bf(wb * vs);
    }
  }
}

// ---------------------------------------------------------------------------
extern "C" void kernel_launch(void* const* d_in, const int* in_sizes, int n_in,
                              void* d_out, int out_size, void* d_ws, size_t ws_size,
                              hipStream_t stream) {
  (void)in_sizes; (void)n_in; (void)out_size; (void)ws_size;
  const float* x   = (const float*)d_in[0];   // (B, N, D)
  const float* Wv  = (const float*)d_in[1];   // (INNER, D)
  const float* Wqk = (const float*)d_in[2];   // (LOGN, 2*INNER, D)
  float* out = (float*)d_out;                 // (B, N, INNER)

  char* ws = (char*)d_ws;
  size_t off = 0;
  auto take = [&](size_t bytes) {
    void* p = ws + off;
    off += (bytes + 255) & ~(size_t)255;
    return p;
  };
  unsigned short* xb   = (unsigned short*)take((size_t)BN_ * D_ * 2);                 // 16 MB
  unsigned short* Wvb  = (unsigned short*)take((size_t)INNER_ * D_ * 2);              // 0.5 MB
  unsigned short* Wqkb = (unsigned short*)take((size_t)LOGN_ * 2 * INNER_ * D_ * 2);  // 12 MB
  float*          vf   = (float*)take((size_t)BN_ * INNER_ * 4);                      // 32 MB
  unsigned short* hA   = (unsigned short*)take((size_t)BN_ * INNER_ * 2);             // 16 MB
  unsigned short* hB   = (unsigned short*)take((size_t)BN_ * INNER_ * 2);             // 16 MB
  unsigned short* qkb  = (unsigned short*)take((size_t)BN_ * 2 * INNER_ * 2);         // 32 MB

  // 1) convert inputs to bf16
  {
    int n4 = BN_ * D_ / 4;
    cvt_f32_bf16_x4<<<(n4 + 255) / 256, 256, 0, stream>>>(x, xb, n4);
    n4 = INNER_ * D_ / 4;
    cvt_f32_bf16_x4<<<(n4 + 255) / 256, 256, 0, stream>>>(Wv, Wvb, n4);
    n4 = LOGN_ * 2 * INNER_ * D_ / 4;
    cvt_f32_bf16_x4<<<(n4 + 255) / 256, 256, 0, stream>>>(Wqk, Wqkb, n4);
  }

  // 2) v = x @ Wv^T  -> fp32 v  +  bf16 h0
  {
    dim3 grid(INNER_ / 128, BN_ / 128);
    gemm_bf16_wmma<<<grid, 256, 0, stream>>>((const __bf16*)xb, (const __bf16*)Wvb,
                                             vf, hA, BN_, INNER_, D_);
  }

  // 3) 12 butterfly stages: qk GEMM (bf16 out) + fused pair-softmax/recombine
  for (int stage = 0; stage < LOGN_; ++stage) {
    const unsigned short* hin = (stage & 1) ? hB : hA;
    unsigned short* hout      = (stage & 1) ? hA : hB;
    const __bf16* Wq = (const __bf16*)(Wqkb + (size_t)stage * 2 * INNER_ * D_);

    dim3 grid(2 * INNER_ / 128, BN_ / 128);
    gemm_bf16_wmma<<<grid, 256, 0, stream>>>((const __bf16*)hin, Wq,
                                             (float*)nullptr, qkb,
                                             BN_, 2 * INNER_, D_);

    const bool last = (stage == LOGN_ - 1);
    stage_pair_softmax<<<(B_ * (N_ / 2) * H_) / 32, 256, 0, stream>>>(
        qkb, vf, last ? (unsigned short*)nullptr : hout, last ? out : (float*)nullptr,
        stage);
  }
}
